// GCN_33208687133526
// MI455X (gfx1250) — matrix operand order repaired and checked by reading, not compiled, not run on hardware
//
#include <hip/hip_runtime.h>

typedef __attribute__((ext_vector_type(16))) _Float16 v16h;
typedef __attribute__((ext_vector_type(8)))  float    v8f;

#define GCN_BN_EPS 1e-5f

// ---------------- degree / normalization ----------------
__global__ __launch_bounds__(256)
void gcn_degree_kernel(const int* __restrict__ src, const int* __restrict__ dst,
                       unsigned* __restrict__ deg_out, unsigned* __restrict__ deg_in, int E) {
  int e = blockIdx.x * blockDim.x + threadIdx.x;
  if (e < E) {
    atomicAdd(&deg_out[src[e]], 1u);
    atomicAdd(&deg_in[dst[e]], 1u);
  }
}

__global__ __launch_bounds__(256)
void gcn_norm_kernel(const unsigned* __restrict__ deg_out, const unsigned* __restrict__ deg_in,
                     float* __restrict__ norm_src, float* __restrict__ norm_dst, int n) {
  int i = blockIdx.x * blockDim.x + threadIdx.x;
  if (i < n) {
    norm_src[i] = rsqrtf(fmaxf((float)deg_out[i], 1.0f));
    norm_dst[i] = rsqrtf(fmaxf((float)deg_in[i], 1.0f));
  }
}

// ------- weight f32 -> f16 packed into WMMA B-fragment layout ---------------
// Fragment (nt, kb) holds B tile K=[kb*32, kb*32+32) x N=[nt*16, nt*16+16).
// Lane L, element j:  n = nt*16 + (L&15), K = kb*32 + (L>>4)*16 + j.
// Stored contiguously: wp[(((nt*4)+kb)*32 + L)*16 + j]  ->  one 32B load/lane.
__global__ __launch_bounds__(256)
void gcn_wpack_kernel(const float* __restrict__ W, _Float16* __restrict__ wp, int ncols) {
  int idx = blockIdx.x * blockDim.x + threadIdx.x;   // total = 128*ncols
  if (idx >= 128 * ncols) return;
  int j    = idx & 15;
  int lane = (idx >> 4) & 31;
  int kb   = (idx >> 9) & 3;
  int nt   = idx >> 11;
  int n = nt * 16 + (lane & 15);
  int K = kb * 32 + (lane >> 4) * 16 + j;
  wp[idx] = (_Float16)W[(size_t)K * ncols + n];
}

// ---------------- edge gather + scatter-add (one wave per edge, 128 feats) ----
// msg = (use_bn ? (h[src]-mean)*invstd : h[src]) * norm_src[src]; agg[dst] += msg
__global__ __launch_bounds__(256)
void gcn_agg_kernel(const float* __restrict__ h, const int* __restrict__ src,
                    const int* __restrict__ dst, const float* __restrict__ norm_src,
                    const float* __restrict__ mean, const float* __restrict__ invstd,
                    float* __restrict__ agg, int E, int use_bn) {
  int wid  = threadIdx.x >> 5;
  int lane = threadIdx.x & 31;
  int e = blockIdx.x * 8 + wid;          // 8 waves per block
  if (e >= E) return;
  int s = src[e], d = dst[e];
  float ns = norm_src[s];
  int c = lane * 4;                      // 32 lanes x float4 = 128 feats
  float4 v = *(const float4*)(h + (size_t)s * 128 + c);
  float vals[4] = {v.x, v.y, v.z, v.w};
  if (use_bn) {
#pragma unroll
    for (int j = 0; j < 4; ++j) vals[j] = (vals[j] - mean[c + j]) * invstd[c + j];
  }
  float* ap = agg + (size_t)d * 128 + c;
#pragma unroll
  for (int j = 0; j < 4; ++j)
    __hip_atomic_fetch_add(&ap[j], vals[j] * ns, __ATOMIC_RELAXED, __HIP_MEMORY_SCOPE_AGENT);
}

// ---------------- per-column batch stats ----------------
__global__ __launch_bounds__(128)
void gcn_stats_kernel(const float* __restrict__ h, float* __restrict__ sum,
                      float* __restrict__ sumsq, int nodes) {
  int col = threadIdx.x;                 // 128 columns
  float s = 0.f, s2 = 0.f;
  for (int r = blockIdx.x; r < nodes; r += gridDim.x) {
    float v = h[(size_t)r * 128 + col];
    s += v; s2 += v * v;
  }
  __hip_atomic_fetch_add(&sum[col],   s,  __ATOMIC_RELAXED, __HIP_MEMORY_SCOPE_AGENT);
  __hip_atomic_fetch_add(&sumsq[col], s2, __ATOMIC_RELAXED, __HIP_MEMORY_SCOPE_AGENT);
}

__global__ __launch_bounds__(128)
void gcn_finalize_stats_kernel(const float* __restrict__ sum, const float* __restrict__ sumsq,
                               float* __restrict__ mean, float* __restrict__ invstd, int nodes) {
  int c = threadIdx.x;
  float m   = sum[c] / (float)nodes;
  float var = sumsq[c] / (float)nodes - m * m;
  mean[c]   = m;
  invstd[c] = rsqrtf(var + GCN_BN_EPS);
}

// ---------------- WMMA matmul: out = relu?( (A*norm_dst) @ W + b ) ----------
// A: [nodes x 128] f32; Wp: pre-packed B fragments (see gcn_wpack_kernel).
// One wave per 16-row slab: A fragments converted to f16 ONCE into registers,
// then reused across all NCOLS/16 n-tiles (W stays L2/L0 resident).
template <int NCOLS, int RELU>
__global__ __launch_bounds__(32)
void gcn_mm_wmma_kernel(const float* __restrict__ A, const v16h* __restrict__ Wp,
                        const float* __restrict__ bias, const float* __restrict__ norm_dst,
                        float* __restrict__ out, int nodes) {
  int lane   = threadIdx.x;
  int lane16 = lane & 15;
  int half   = lane >> 4;                // K-half select (contiguous k order, matches Wp)
  int mt = blockIdx.x;

  int r  = mt * 16 + lane16;
  int rc = (r < nodes) ? r : (nodes - 1); // clamp: keep EXEC all-ones for WMMA
  float scale = norm_dst[rc];
  const float* arow = A + (size_t)rc * 128;
  const int baseA = half * 16;           // contiguous k order: lanes16-31 hold K+16

  // Preload + convert all 4 A fragments (K=128) into registers.
  v16h afrag[4];
#pragma unroll
  for (int kb = 0; kb < 4; ++kb) {
    const float* p0 = arow + kb * 32 + baseA;        // K = kk+baseA .. +7
    const float* p1 = arow + kb * 32 + 8 + baseA;    // K = kk+8+baseA .. +7
#pragma unroll
    for (int i = 0; i < 8; ++i) {
      afrag[kb][i]     = (_Float16)(p0[i] * scale);
      afrag[kb][i + 8] = (_Float16)(p1[i] * scale);
    }
  }

  const bool full_tile = (mt * 16 + 16) <= nodes;   // uniform per wave

#pragma unroll
  for (int nt = 0; nt < NCOLS / 16; ++nt) {
    const v16h* wfrag = Wp + (size_t)(nt * 4) * 32 + lane;  // fragment stride 32 v16h
    v8f c = {};
#pragma unroll
    for (int kb = 0; kb < 4; ++kb)
      c = __builtin_amdgcn_wmma_f32_16x16x32_f16(false, afrag[kb], false,
                                                 wfrag[kb * 32], (short)0, c,
                                                 false, false);
    int n = nt * 16 + lane16;
    float bn = bias[n];
    float res[8];
#pragma unroll
    for (int p = 0; p < 8; ++p) {
      float v = c[p] + bn;
      res[p] = RELU ? fmaxf(v, 0.f) : v;
    }
    // C/D layout: VGPR p -> M = mt*16 + half*8 + p; stores share one base.
    float* op = out + (size_t)(mt * 16 + half * 8) * NCOLS + n;
    if (full_tile) {
#pragma unroll
      for (int p = 0; p < 8; ++p) op[(size_t)p * NCOLS] = res[p];
    } else {
#pragma unroll
      for (int p = 0; p < 8; ++p)
        if (mt * 16 + half * 8 + p < nodes) op[(size_t)p * NCOLS] = res[p];
    }
  }
}

// ---------------- host orchestration ----------------
extern "C" void kernel_launch(void* const* d_in, const int* in_sizes, int n_in,
                              void* d_out, int out_size, void* d_ws, size_t ws_size,
                              hipStream_t stream) {
  const float* feats = (const float*)d_in[0];
  const float* W0 = (const float*)d_in[1];
  const float* b0 = (const float*)d_in[2];
  const float* W1 = (const float*)d_in[3];
  const float* b1 = (const float*)d_in[4];
  const float* W2 = (const float*)d_in[5];
  const float* b2 = (const float*)d_in[6];
  const int*   src = (const int*)d_in[7];
  const int*   dst = (const int*)d_in[8];

  const int IN = 128, H = 128, C = 64;
  const int nodes = in_sizes[0] / IN;
  const int E     = in_sizes[7];
  float* out = (float*)d_out;

  // workspace carve-up (256B aligned)
  char* ws = (char*)d_ws;
  size_t off = 0;
  auto carve = [&](size_t bytes) -> void* {
    void* p = ws + off;
    off += (bytes + 255) & ~(size_t)255;
    return p;
  };
  unsigned* deg_out  = (unsigned*)carve((size_t)nodes * 4);
  unsigned* deg_in   = (unsigned*)carve((size_t)nodes * 4);
  float*    norm_src = (float*)   carve((size_t)nodes * 4);
  float*    norm_dst = (float*)   carve((size_t)nodes * 4);
  _Float16* W0p = (_Float16*)carve((size_t)IN * H * 2);   // packed fragments
  _Float16* W1p = (_Float16*)carve((size_t)H * H * 2);
  _Float16* W2p = (_Float16*)carve((size_t)H * C * 2);
  float* sum    = (float*)carve(128 * 4);
  float* sumsq  = (float*)carve(128 * 4);
  float* mean   = (float*)carve(128 * 4);
  float* invstd = (float*)carve(128 * 4);
  float* bufA = (float*)carve((size_t)nodes * 128 * 4);  // aggregation target
  float* bufB = (float*)carve((size_t)nodes * 128 * 4);  // layer activations

  const int tilesM = (nodes + 15) / 16;
  dim3 aggGrid((unsigned)((E + 7) / 8));

  // degrees -> norms
  hipMemsetAsync(deg_out, 0, (size_t)nodes * 4, stream);
  hipMemsetAsync(deg_in,  0, (size_t)nodes * 4, stream);
  gcn_degree_kernel<<<(E + 255) / 256, 256, 0, stream>>>(src, dst, deg_out, deg_in, E);
  gcn_norm_kernel<<<(nodes + 255) / 256, 256, 0, stream>>>(deg_out, deg_in,
                                                           norm_src, norm_dst, nodes);
  // weights -> packed f16 WMMA B fragments (once per launch)
  gcn_wpack_kernel<<<(IN * H + 255) / 256, 256, 0, stream>>>(W0, W0p, H);
  gcn_wpack_kernel<<<(H * H + 255) / 256, 256, 0, stream>>>(W1, W1p, H);
  gcn_wpack_kernel<<<(H * C + 255) / 256, 256, 0, stream>>>(W2, W2p, C);

  // ---- layer 0: agg(features) -> mm(W0)+relu ----
  hipMemsetAsync(bufA, 0, (size_t)nodes * 128 * 4, stream);
  gcn_agg_kernel<<<aggGrid, 256, 0, stream>>>(feats, src, dst, norm_src,
                                              mean, invstd, bufA, E, 0);
  gcn_mm_wmma_kernel<128, 1><<<tilesM, 32, 0, stream>>>(
      bufA, (const v16h*)W0p, b0, norm_dst, bufB, nodes);

  // ---- BN1 stats (applied fused into next gather) ----
  hipMemsetAsync(sum,   0, 128 * 4, stream);
  hipMemsetAsync(sumsq, 0, 128 * 4, stream);
  gcn_stats_kernel<<<1024, 128, 0, stream>>>(bufB, sum, sumsq, nodes);
  gcn_finalize_stats_kernel<<<1, 128, 0, stream>>>(sum, sumsq, mean, invstd, nodes);

  // ---- layer 1: agg(BN(h1)) -> mm(W1)+relu ----
  hipMemsetAsync(bufA, 0, (size_t)nodes * 128 * 4, stream);
  gcn_agg_kernel<<<aggGrid, 256, 0, stream>>>(bufB, src, dst, norm_src,
                                              mean, invstd, bufA, E, 1);
  gcn_mm_wmma_kernel<128, 1><<<tilesM, 32, 0, stream>>>(
      bufA, (const v16h*)W1p, b1, norm_dst, bufB, nodes);

  // ---- BN2 stats ----
  hipMemsetAsync(sum,   0, 128 * 4, stream);
  hipMemsetAsync(sumsq, 0, 128 * 4, stream);
  gcn_stats_kernel<<<1024, 128, 0, stream>>>(bufB, sum, sumsq, nodes);
  gcn_finalize_stats_kernel<<<1, 128, 0, stream>>>(sum, sumsq, mean, invstd, nodes);

  // ---- layer 2: agg(BN(h2)) -> mm(W2), no relu, out [nodes x 64] ----
  hipMemsetAsync(bufA, 0, (size_t)nodes * 128 * 4, stream);
  gcn_agg_kernel<<<aggGrid, 256, 0, stream>>>(bufB, src, dst, norm_src,
                                              mean, invstd, bufA, E, 1);
  gcn_mm_wmma_kernel<64, 0><<<tilesM, 32, 0, stream>>>(
      bufA, (const v16h*)W2p, b2, norm_dst, out, nodes);
}
